// LoRACrossAttnProcessor_77077483094696
// MI455X (gfx1250) — matrix operand order repaired and checked by reading, not compiled, not run on hardware
//
#include <hip/hip_runtime.h>
#include <hip/hip_bf16.h>

#define HH   20
#define DD   64
#define HID  1280
#define CTX  2048
#define RNK  4
#define BQ   8
#define SQ   4096
#define TQ   77
#define SCALEF 0.125f
#define LSC  1.0f
#define FSC  1.0f

typedef __attribute__((ext_vector_type(16))) __bf16 v16bf;
typedef __attribute__((ext_vector_type(8)))  float  v8f;

union Frag { unsigned int u[8]; v16bf v; };

__device__ __forceinline__ unsigned short f2bf(float f) {
  union { float f; unsigned int u; } x; x.f = f;
  unsigned int r = x.u + 0x7FFFu + ((x.u >> 16) & 1u);
  return (unsigned short)(r >> 16);
}

__device__ __forceinline__ int a_k(int p, int half) {
  // 16-bit A-matrix 16x32 layout: VGPR p<4 -> K=2p,2p+1 ; p>=4 -> K=16+2(p-4) ; +8 for hi half
  return ((p < 4) ? (2 * p) : (16 + 2 * (p - 4))) + half * 8;
}

__device__ __forceinline__ v8f wmma_bf16(const Frag& a, const Frag& b, v8f c) {
  return __builtin_amdgcn_wmma_f32_16x16x32_bf16(false, a.v, false, b.v, (short)0, c, false, false);
}

// CDNA5 async copy: 16 bytes per lane, global -> LDS, tracked by ASYNCcnt.
__device__ __forceinline__ void async_copy16(unsigned int lds_off, const void* gaddr) {
  asm volatile("global_load_async_to_lds_b128 %0, %1, off"
               :: "v"(lds_off), "v"(gaddr) : "memory");
}
__device__ __forceinline__ void wait_async0() {
  asm volatile("s_wait_asynccnt 0x0" ::: "memory");
}

// ---------------- f32 -> bf16 (packed pairs) ----------------
__global__ void cvt_bf16(const float* __restrict__ in, unsigned int* __restrict__ out, int n2) {
  int i = blockIdx.x * blockDim.x + threadIdx.x;
  if (i < n2) {
    float a = in[2 * i], b = in[2 * i + 1];
    out[i] = (unsigned int)f2bf(a) | ((unsigned int)f2bf(b) << 16);
  }
}

// ---------------- W' = W + LS * down @ up  (bf16 out) ----------------
__global__ void fuse_w(const float* __restrict__ W, const float* __restrict__ dn,
                       const float* __restrict__ up, unsigned short* __restrict__ out,
                       int K, int N) {
  int i = blockIdx.x * blockDim.x + threadIdx.x;
  if (i >= K * N) return;
  int k = i / N, n = i - k * N;
  float acc = W[i];
#pragma unroll
  for (int r = 0; r < RNK; ++r) acc += LSC * dn[k * RNK + r] * up[r * N + n];
  out[i] = f2bf(acc);
}

// ---------------- bf16 GEMM: C[M,N] = A[M,K] @ B[K,N] ----------------
// Double-buffered LDS; A staged via GLOBAL_LOAD_ASYNC_TO_LDS_B128, B prefetched to
// registers and scatter-transposed to LDS, both overlapped with WMMA on the other buffer.
// mode 0: store bf16 to Cb.   mode 1: store f32 to Cf + bias[n] + resid[row,n].
__launch_bounds__(256)
__global__ void gemm_bf16(const unsigned short* __restrict__ A,
                          const unsigned short* __restrict__ B,
                          int M, int K, int N,
                          unsigned short* __restrict__ Cb,
                          float* __restrict__ Cf,
                          const float* __restrict__ bias,
                          const float* __restrict__ resid,
                          int mode) {
  __shared__ unsigned short As[2][128][40];   // [buf][row][k]
  __shared__ unsigned short Bs[2][128][40];   // [buf][n][k] (transposed)
  const int t = threadIdx.x;
  const int lane = t & 31, wave = t >> 5;
  const int lr = lane & 15, half = lane >> 4;
  const int m0 = blockIdx.y * 128, n0 = blockIdx.x * 128;
  const int wm = (wave & 1) * 64, wn = (wave >> 1) * 32;
  const int nk = K >> 5;

  // per-thread staging coordinates (2 x 16B chunks each for A and B tiles)
  int aRow[2], aKk[2], bKr[2], bNn[2];
#pragma unroll
  for (int i = 0; i < 2; ++i) {
    int e = (t + i * 256) * 8;
    aRow[i] = e >> 5;  aKk[i] = e & 31;
    bKr[i] = e >> 7;   bNn[i] = e & 127;
  }

  auto issueA = [&](int kt, int buf) {
#pragma unroll
    for (int i = 0; i < 2; ++i) {
      const unsigned short* g = &A[(size_t)(m0 + aRow[i]) * K + kt * 32 + aKk[i]];
      unsigned int lds = (unsigned int)(size_t)&As[buf][aRow[i]][aKk[i]];
      async_copy16(lds, (const void*)g);
    }
  };
  auto loadB = [&](int kt, uint4* v) {
#pragma unroll
    for (int i = 0; i < 2; ++i)
      v[i] = *(const uint4*)&B[(size_t)(kt * 32 + bKr[i]) * N + n0 + bNn[i]];
  };
  auto storeB = [&](int buf, const uint4* v) {
#pragma unroll
    for (int i = 0; i < 2; ++i) {
      unsigned int w0 = v[i].x, w1 = v[i].y, w2 = v[i].z, w3 = v[i].w;
      int nn = bNn[i], kr = bKr[i];
      Bs[buf][nn + 0][kr] = (unsigned short)(w0);
      Bs[buf][nn + 1][kr] = (unsigned short)(w0 >> 16);
      Bs[buf][nn + 2][kr] = (unsigned short)(w1);
      Bs[buf][nn + 3][kr] = (unsigned short)(w1 >> 16);
      Bs[buf][nn + 4][kr] = (unsigned short)(w2);
      Bs[buf][nn + 5][kr] = (unsigned short)(w2 >> 16);
      Bs[buf][nn + 6][kr] = (unsigned short)(w3);
      Bs[buf][nn + 7][kr] = (unsigned short)(w3 >> 16);
    }
  };

  v8f acc[4][2];
  const v8f z8 = {0.f, 0.f, 0.f, 0.f, 0.f, 0.f, 0.f, 0.f};
#pragma unroll
  for (int i = 0; i < 4; ++i)
#pragma unroll
    for (int j = 0; j < 2; ++j) acc[i][j] = z8;

  uint4 breg[2];
  // prologue: stage tile 0 into buffer 0
  issueA(0, 0);
  loadB(0, breg);
  storeB(0, breg);
  wait_async0();
  __syncthreads();

  for (int kt = 0; kt < nk; ++kt) {
    const int cur = kt & 1, nxt = cur ^ 1;
    const bool more = (kt + 1) < nk;
    if (more) {
      issueA(kt + 1, nxt);   // async global->LDS, overlapped with compute below
      loadB(kt + 1, breg);   // global->regs, overlapped with compute below
    }

    Frag af[4], bf_[2];
#pragma unroll
    for (int p = 0; p < 8; ++p) {
      int ka = a_k(p, half);
#pragma unroll
      for (int ms = 0; ms < 4; ++ms)
        af[ms].u[p] = *(const unsigned int*)&As[cur][wm + ms * 16 + lr][ka];
      int kb = a_k(p, half);  // B 32x16: same K layout as A, N=lane
#pragma unroll
      for (int ns = 0; ns < 2; ++ns)
        bf_[ns].u[p] = *(const unsigned int*)&Bs[cur][wn + ns * 16 + lr][kb];
    }
#pragma unroll
    for (int ms = 0; ms < 4; ++ms)
#pragma unroll
      for (int ns = 0; ns < 2; ++ns)
        acc[ms][ns] = wmma_bf16(af[ms], bf_[ns], acc[ms][ns]);

    if (more) {
      storeB(nxt, breg);   // Bs[nxt] last read in kt-1, protected by that barrier
      wait_async0();       // As[nxt] async writes complete before signaling
    }
    __syncthreads();
  }

#pragma unroll
  for (int ms = 0; ms < 4; ++ms)
#pragma unroll
    for (int ns = 0; ns < 2; ++ns)
#pragma unroll
      for (int r = 0; r < 8; ++r) {
        int row = m0 + wm + ms * 16 + r + half * 8;
        int col = n0 + wn + ns * 16 + lr;
        if (row < M) {
          float v = acc[ms][ns][r];
          if (mode == 0)
            Cb[(size_t)row * N + col] = f2bf(v);
          else
            Cf[(size_t)row * N + col] = v + bias[col] + resid[(size_t)row * N + col];
        }
      }
}

// ---------------- fused cross-attention ----------------
// keys 0..76 text, 77..79 zero pad, 80..95 image. grid=(S/128, H, B), 256 thr.
__launch_bounds__(256)
__global__ void cross_attn(const unsigned short* __restrict__ qb,
                           const unsigned short* __restrict__ kb,
                           const unsigned short* __restrict__ vb,
                           const unsigned short* __restrict__ ikb,
                           const unsigned short* __restrict__ ivb,
                           const int* __restrict__ maskp,
                           unsigned short* __restrict__ ob) {
  __shared__ unsigned short Ks[96][64];     // [key][d]
  __shared__ unsigned short Vs[64][96];     // transposed: [d][key]
  __shared__ unsigned short Ps[8][16][96];  // per-wave probs [row][key]
  const int t = threadIdx.x;
  const int lane = t & 31, wave = t >> 5;
  const int lr = lane & 15, half = lane >> 4;
  const int qs = blockIdx.x, h = blockIdx.y, b = blockIdx.z;

  for (int idx = t; idx < 96 * 32; idx += 256) {
    int key = idx >> 5;
    int dp = (idx & 31) * 2;
    unsigned int kv = 0u, vv = 0u;
    if (key < 77) {
      size_t base = (size_t)(b * TQ + key) * HID + h * DD + dp;
      kv = *(const unsigned int*)&kb[base];
      vv = *(const unsigned int*)&vb[base];
    } else if (key >= 80) {
      size_t base = (size_t)(b * 16 + (key - 80)) * HID + h * DD + dp;
      kv = *(const unsigned int*)&ikb[base];
      vv = *(const unsigned int*)&ivb[base];
    }
    *(unsigned int*)&Ks[key][dp] = kv;
    Vs[dp + 0][key] = (unsigned short)(vv & 0xffffu);
    Vs[dp + 1][key] = (unsigned short)(vv >> 16);
  }
  __syncthreads();

  int mv0 = maskp[b * 4 + 0], mv1 = maskp[b * 4 + 1];
  int mv2 = maskp[b * 4 + 2], mv3 = maskp[b * 4 + 3];
  bool anyv = (mv0 | mv1 | mv2 | mv3) != 0;
  int obj = lr >> 2;
  int objm = (obj == 0) ? mv0 : (obj == 1) ? mv1 : (obj == 2) ? mv2 : mv3;
  float addi = anyv ? (objm ? 0.f : -1e6f) : 0.f;
  float iscale = anyv ? FSC : 0.f;

  const int srow = qs * 128 + wave * 16 + lr;
  const v8f z8 = {0.f, 0.f, 0.f, 0.f, 0.f, 0.f, 0.f, 0.f};

  v8f sacc[6];
#pragma unroll
  for (int i = 0; i < 6; ++i) sacc[i] = z8;

#pragma unroll
  for (int kk = 0; kk < 2; ++kk) {
    Frag qa;
#pragma unroll
    for (int p = 0; p < 8; ++p) {
      int ka = kk * 32 + a_k(p, half);
      qa.u[p] = *(const unsigned int*)&qb[(size_t)(b * SQ + srow) * HID + h * DD + ka];
    }
#pragma unroll
    for (int kt = 0; kt < 6; ++kt) {
      Frag kf;
#pragma unroll
      for (int v = 0; v < 8; ++v) {
        int kd = kk * 32 + a_k(v, half);
        kf.u[v] = *(const unsigned int*)&Ks[kt * 16 + lr][kd];
      }
      sacc[kt] = wmma_bf16(qa, kf, sacc[kt]);
    }
  }

  // dual softmax (text tiles 0..4 masked past key 77; image tile 5 with validity mask)
#pragma unroll
  for (int r = 0; r < 8; ++r) {
    float mx = -1e30f;
#pragma unroll
    for (int kt = 0; kt < 5; ++kt) {
      float s = sacc[kt][r] * SCALEF;
      if (kt * 16 + lr >= 77) s = -1e30f;
      sacc[kt][r] = s;
      mx = fmaxf(mx, s);
    }
    mx = fmaxf(mx, __shfl_xor(mx, 1));
    mx = fmaxf(mx, __shfl_xor(mx, 2));
    mx = fmaxf(mx, __shfl_xor(mx, 4));
    mx = fmaxf(mx, __shfl_xor(mx, 8));
    float sum = 0.f;
#pragma unroll
    for (int kt = 0; kt < 5; ++kt) {
      float e = __expf(sacc[kt][r] - mx);
      sacc[kt][r] = e;
      sum += e;
    }
    sum += __shfl_xor(sum, 1);
    sum += __shfl_xor(sum, 2);
    sum += __shfl_xor(sum, 4);
    sum += __shfl_xor(sum, 8);
    float inv = 1.f / sum;
#pragma unroll
    for (int kt = 0; kt < 5; ++kt) sacc[kt][r] *= inv;

    float si = sacc[5][r] * SCALEF + addi;
    float mi = si;
    mi = fmaxf(mi, __shfl_xor(mi, 1));
    mi = fmaxf(mi, __shfl_xor(mi, 2));
    mi = fmaxf(mi, __shfl_xor(mi, 4));
    mi = fmaxf(mi, __shfl_xor(mi, 8));
    float ei = __expf(si - mi);
    float ss = ei;
    ss += __shfl_xor(ss, 1);
    ss += __shfl_xor(ss, 2);
    ss += __shfl_xor(ss, 4);
    ss += __shfl_xor(ss, 8);
    sacc[5][r] = ei * (iscale / ss);
  }

  // write probs to per-wave LDS region (C-layout -> A-layout re-staging)
#pragma unroll
  for (int kt = 0; kt < 6; ++kt)
#pragma unroll
    for (int r = 0; r < 8; ++r)
      Ps[wave][r + half * 8][kt * 16 + lr] = f2bf(sacc[kt][r]);

  v8f oacc[4];
#pragma unroll
  for (int i = 0; i < 4; ++i) oacc[i] = z8;

#pragma unroll
  for (int ks = 0; ks < 3; ++ks) {
    Frag pa;
#pragma unroll
    for (int p = 0; p < 8; ++p) {
      int ka = ks * 32 + a_k(p, half);
      pa.u[p] = *(const unsigned int*)&Ps[wave][lr][ka];
    }
#pragma unroll
    for (int dt = 0; dt < 4; ++dt) {
      Frag vf;
#pragma unroll
      for (int v = 0; v < 8; ++v) {
        int key = ks * 32 + a_k(v, half);
        vf.u[v] = *(const unsigned int*)&Vs[dt * 16 + lr][key];
      }
      oacc[dt] = wmma_bf16(pa, vf, oacc[dt]);
    }
  }

#pragma unroll
  for (int dt = 0; dt < 4; ++dt)
#pragma unroll
    for (int r = 0; r < 8; ++r) {
      int row = qs * 128 + wave * 16 + r + half * 8;
      ob[(size_t)(b * SQ + row) * HID + h * DD + dt * 16 + lr] = f2bf(oacc[dt][r]);
    }
}

// ---------------- host ----------------
extern "C" void kernel_launch(void* const* d_in, const int* in_sizes, int n_in,
                              void* d_out, int out_size, void* d_ws, size_t ws_size,
                              hipStream_t stream) {
  const float* hs  = (const float*)d_in[0];
  const float* ehs = (const float*)d_in[1];
  const float* ihs = (const float*)d_in[2];
  const int*   msk = (const int*)d_in[3];
  const float* Wq  = (const float*)d_in[4];
  const float* Wk  = (const float*)d_in[5];
  const float* Wv  = (const float*)d_in[6];
  const float* Wo  = (const float*)d_in[7];
  const float* bo  = (const float*)d_in[8];
  const float* qd  = (const float*)d_in[9];
  const float* qu  = (const float*)d_in[10];
  const float* kd  = (const float*)d_in[11];
  const float* ku  = (const float*)d_in[12];
  const float* vd  = (const float*)d_in[13];
  const float* vu  = (const float*)d_in[14];
  const float* kid = (const float*)d_in[15];
  const float* kiu = (const float*)d_in[16];
  const float* vid = (const float*)d_in[17];
  const float* viu = (const float*)d_in[18];
  const float* od  = (const float*)d_in[19];
  const float* ou  = (const float*)d_in[20];
  float* outp = (float*)d_out;

  char* ws = (char*)d_ws;
  size_t off = 0;
  auto take = [&](size_t elems) -> unsigned short* {
    unsigned short* p = (unsigned short*)(ws + off);
    off += (elems * 2 + 255) & ~(size_t)255;
    return p;
  };
  const size_t MQ = (size_t)BQ * SQ;  // 32768
  unsigned short* hsb   = take(MQ * HID);
  unsigned short* ehsb  = take((size_t)BQ * TQ * CTX);
  unsigned short* ihsb  = take((size_t)BQ * 16 * CTX);
  unsigned short* Wqf   = take((size_t)HID * HID);
  unsigned short* Wkf   = take((size_t)CTX * HID);
  unsigned short* Wvf   = take((size_t)CTX * HID);
  unsigned short* Wkif  = take((size_t)CTX * HID);
  unsigned short* Wvif  = take((size_t)CTX * HID);
  unsigned short* Wof   = take((size_t)HID * HID);
  unsigned short* qbuf  = take(MQ * HID);
  unsigned short* kbuf  = take((size_t)BQ * TQ * HID);
  unsigned short* vbuf  = take((size_t)BQ * TQ * HID);
  unsigned short* ikbuf = take((size_t)BQ * 16 * HID);
  unsigned short* ivbuf = take((size_t)BQ * 16 * HID);
  unsigned short* attnb = take(MQ * HID);

  // activations -> bf16
  {
    int n2 = (int)(MQ * HID / 2);
    cvt_bf16<<<(n2 + 255) / 256, 256, 0, stream>>>(hs, (unsigned int*)hsb, n2);
    n2 = BQ * TQ * CTX / 2;
    cvt_bf16<<<(n2 + 255) / 256, 256, 0, stream>>>(ehs, (unsigned int*)ehsb, n2);
    n2 = BQ * 16 * CTX / 2;
    cvt_bf16<<<(n2 + 255) / 256, 256, 0, stream>>>(ihs, (unsigned int*)ihsb, n2);
  }
  // fold LoRA into weights (bf16)
  fuse_w<<<(HID * HID + 255) / 256, 256, 0, stream>>>(Wq, qd, qu, Wqf, HID, HID);
  fuse_w<<<(CTX * HID + 255) / 256, 256, 0, stream>>>(Wk, kd, ku, Wkf, CTX, HID);
  fuse_w<<<(CTX * HID + 255) / 256, 256, 0, stream>>>(Wv, vd, vu, Wvf, CTX, HID);
  fuse_w<<<(CTX * HID + 255) / 256, 256, 0, stream>>>(Wk, kid, kiu, Wkif, CTX, HID);
  fuse_w<<<(CTX * HID + 255) / 256, 256, 0, stream>>>(Wv, vid, viu, Wvif, CTX, HID);
  fuse_w<<<(HID * HID + 255) / 256, 256, 0, stream>>>(Wo, od, ou, Wof, HID, HID);

  dim3 blk(256);
  // projections
  gemm_bf16<<<dim3(10, 256), blk, 0, stream>>>(hsb, Wqf, (int)MQ, HID, HID, qbuf, nullptr, nullptr, nullptr, 0);
  gemm_bf16<<<dim3(10, 5),   blk, 0, stream>>>(ehsb, Wkf, BQ * TQ, CTX, HID, kbuf, nullptr, nullptr, nullptr, 0);
  gemm_bf16<<<dim3(10, 5),   blk, 0, stream>>>(ehsb, Wvf, BQ * TQ, CTX, HID, vbuf, nullptr, nullptr, nullptr, 0);
  gemm_bf16<<<dim3(10, 1),   blk, 0, stream>>>(ihsb, Wkif, BQ * 16, CTX, HID, ikbuf, nullptr, nullptr, nullptr, 0);
  gemm_bf16<<<dim3(10, 1),   blk, 0, stream>>>(ihsb, Wvif, BQ * 16, CTX, HID, ivbuf, nullptr, nullptr, nullptr, 0);

  // fused text+image attention
  cross_attn<<<dim3(SQ / 128, HH, BQ), blk, 0, stream>>>(qbuf, kbuf, vbuf, ikbuf, ivbuf, msk, attnb);

  // output projection + bias + residual -> f32 out
  gemm_bf16<<<dim3(10, 256), blk, 0, stream>>>(attnb, Wof, (int)MQ, HID, HID, nullptr, outp, bo, hs, 1);
}
